// cons_kd_43765716747000
// MI455X (gfx1250) — matrix operand order repaired and checked by reading, not compiled, not run on hardware
//
#include <hip/hip_runtime.h>
#include <hip/hip_bf16.h>

// ---------------------------------------------------------------------------
// CDNA5 (gfx1250) implementation of the 5-token transformer pipeline.
// Dense GEMMs: v_wmma_f32_16x16x32_bf16, 2x2 register blocking per wave,
// async-to-LDS A staging (global_load_async_to_lds_b128 / s_wait_asynccnt),
// double-buffered 32x128 slabs, weight prefetch one slab ahead.
// ---------------------------------------------------------------------------

typedef __attribute__((ext_vector_type(16))) __bf16 v16bf;
typedef __attribute__((ext_vector_type(8)))  float  v8f;

union FragAB { uint4 u[2]; v16bf v; };

__device__ __forceinline__ unsigned short f2bf(float x) {
    unsigned u = __float_as_uint(x);
    unsigned r = u + 0x7FFFu + ((u >> 16) & 1u);   // round-to-nearest-even
    return (unsigned short)(r >> 16);
}
__device__ __forceinline__ float bf2f(unsigned short b) {
    return __uint_as_float(((unsigned)b) << 16);
}

// ---------------------------------------------------------------------------
// fp32 -> bf16 weight conversion
// ---------------------------------------------------------------------------
__global__ void k_cvt(const float* __restrict__ s, unsigned short* __restrict__ d, int n) {
    int i = blockIdx.x * blockDim.x + threadIdx.x;
    if (i < n) d[i] = f2bf(s[i]);
}

// ---------------------------------------------------------------------------
// argmin over 5 distances + build merge1 input [B,384] = cat(obs, gather)
// ---------------------------------------------------------------------------
__global__ __launch_bounds__(128) void k_prep(
    const float* __restrict__ obs, const float* __restrict__ last_hid,
    const float* __restrict__ dis, int* __restrict__ idxb,
    unsigned short* __restrict__ A1)
{
    int b = blockIdx.x, t = threadIdx.x;
    int idx = 0; float best = dis[(size_t)b * 5];
    #pragma unroll
    for (int n = 1; n < 5; ++n) {
        float dn = dis[(size_t)b * 5 + n];
        if (dn < best) { best = dn; idx = n; }
    }
    if (t == 0) idxb[b] = idx;
    for (int c = t; c < 384; c += 128) {
        float val = (c < 128) ? obs[(size_t)b * 128 + c]
                              : last_hid[((size_t)b * 5 + idx) * 256 + (c - 128)];
        A1[(size_t)b * 384 + c] = f2bf(val);
    }
}

// ---------------------------------------------------------------------------
// Generic WMMA GEMM: C[M,N] = A[M,K](bf16) x W[N,K]^T(bf16) + bias, opt relu.
// Block = 256 thr (8 waves); block tile 32 rows x 256 cols.
// Each wave: 2 M-tiles x 2 N-tiles (4 WMMA/k-step, B frags reused across M).
// A staged in 32x128 slabs via global_load_async_to_lds_b128, double-buffered:
// one barrier per 4 WMMA k-steps, DMA of slab s+1 overlaps compute of slab s.
// Requires: M % 32 == 0, N % 256 == 0, K % 128 == 0 (all shapes here qualify).
// ---------------------------------------------------------------------------
#define SLAB_K   128
#define LDS_ROW  136                  // 128 + 8 pad (17x16B rows: conflict-free b128)

__global__ __launch_bounds__(256) void k_gemm(
    const unsigned short* __restrict__ A, const unsigned short* __restrict__ W,
    const float* __restrict__ bias, float* __restrict__ Cf,
    unsigned short* __restrict__ Cb, int M, int N, int K, int relu)
{
    __shared__ unsigned short ldsA[2][32 * LDS_ROW];
    const int mtiles = M >> 5;                    // 32-row block tiles
    const int mt     = blockIdx.x % mtiles;
    const int nbase  = (blockIdx.x / mtiles) << 8;
    const int w      = threadIdx.x >> 5;
    const int l      = threadIdx.x & 31;
    const int half   = l >> 4;
    const int lane16 = l & 15;
    const int c0     = nbase + (w << 5) + lane16;
    const int c1     = c0 + 16;

    v8f acc00 = {0.f,0.f,0.f,0.f,0.f,0.f,0.f,0.f};
    v8f acc01 = acc00, acc10 = acc00, acc11 = acc00;

    // staging: thread t moves rows (t>>4) and (t>>4)+16, chunk (t&15): 2x16B
    const int r_st = threadIdx.x >> 4;
    const int c_st = (threadIdx.x & 15) * 8;
    const unsigned short* gA0 = A + (size_t)(mt * 32 + r_st) * K + c_st;
    const unsigned short* gA1 = A + (size_t)(mt * 32 + r_st + 16) * K + c_st;
    const unsigned st0a = (unsigned)(uintptr_t)&ldsA[0][r_st * LDS_ROW + c_st];
    const unsigned st0b = (unsigned)(uintptr_t)&ldsA[0][(r_st + 16) * LDS_ROW + c_st];
    const unsigned st1a = (unsigned)(uintptr_t)&ldsA[1][r_st * LDS_ROW + c_st];
    const unsigned st1b = (unsigned)(uintptr_t)&ldsA[1][(r_st + 16) * LDS_ROW + c_st];

    const size_t b0row = (size_t)c0 * K + half * 16;
    const size_t b1row = (size_t)c1 * K + half * 16;
    const int nslab = K / SLAB_K;

    // stage slab 0 into buffer 0 (async DMA, no VGPR round trip)
    asm volatile("global_load_async_to_lds_b128 %0, %1, off"
                 :: "v"(st0a), "v"(gA0) : "memory");
    asm volatile("global_load_async_to_lds_b128 %0, %1, off"
                 :: "v"(st0b), "v"(gA1) : "memory");
    asm volatile("s_wait_asynccnt 0x0" ::: "memory");
    __syncthreads();

    for (int s = 0; s < nslab; ++s) {
        const int cur = s & 1;
        // kick off DMA of the next slab into the other buffer
        if (s + 1 < nslab) {
            const int ko = (s + 1) * SLAB_K;
            asm volatile("global_load_async_to_lds_b128 %0, %1, off"
                         :: "v"(cur ? st0a : st1a), "v"(gA0 + ko) : "memory");
            asm volatile("global_load_async_to_lds_b128 %0, %1, off"
                         :: "v"(cur ? st0b : st1b), "v"(gA1 + ko) : "memory");
            // prefetch next-slab weights for both N tiles toward the WGP
            __builtin_prefetch(W + b0row + ko, 0, 3);
            __builtin_prefetch(W + b1row + ko, 0, 3);
        }
        // 4 WMMA k-steps from the current slab; B frags reused across 2 M-tiles
        #pragma unroll
        for (int ks = 0; ks < SLAB_K / 32; ++ks) {
            FragAB fa0, fa1, fb0, fb1;
            const unsigned short* ap = &ldsA[cur][lane16 * LDS_ROW + ks * 32 + half * 16];
            fa0.u[0] = *(const uint4*)(ap);
            fa0.u[1] = *(const uint4*)(ap + 8);
            fa1.u[0] = *(const uint4*)(ap + 16 * LDS_ROW);
            fa1.u[1] = *(const uint4*)(ap + 16 * LDS_ROW + 8);
            const int kb = s * SLAB_K + ks * 32;
            const uint4* wp0 = (const uint4*)(W + b0row + kb);
            fb0.u[0] = wp0[0]; fb0.u[1] = wp0[1];
            const uint4* wp1 = (const uint4*)(W + b1row + kb);
            fb1.u[0] = wp1[0]; fb1.u[1] = wp1[1];
            acc00 = __builtin_amdgcn_wmma_f32_16x16x32_bf16(false, fa0.v, false, fb0.v,
                                                            (short)0, acc00, false, false);
            acc01 = __builtin_amdgcn_wmma_f32_16x16x32_bf16(false, fa0.v, false, fb1.v,
                                                            (short)0, acc01, false, false);
            acc10 = __builtin_amdgcn_wmma_f32_16x16x32_bf16(false, fa1.v, false, fb0.v,
                                                            (short)0, acc10, false, false);
            acc11 = __builtin_amdgcn_wmma_f32_16x16x32_bf16(false, fa1.v, false, fb1.v,
                                                            (short)0, acc11, false, false);
        }
        // drain this wave's DMA, then joint barrier:
        // (a) slab s+1 visible to all, (b) slab s fully consumed before reuse
        asm volatile("s_wait_asynccnt 0x0" ::: "memory");
        __syncthreads();
    }

    const float bias0 = bias[c0], bias1 = bias[c1];
    #pragma unroll
    for (int v = 0; v < 8; ++v) {
        const int rg0 = mt * 32 + v + half * 8;   // M-tile 0 rows
        const int rg1 = rg0 + 16;                 // M-tile 1 rows
        const size_t o00 = (size_t)rg0 * N + c0;
        const size_t o01 = (size_t)rg0 * N + c1;
        const size_t o10 = (size_t)rg1 * N + c0;
        const size_t o11 = (size_t)rg1 * N + c1;
        float x00 = acc00[v] + bias0;
        float x01 = acc01[v] + bias1;
        float x10 = acc10[v] + bias0;
        float x11 = acc11[v] + bias1;
        if (relu) {
            x00 = fmaxf(x00, 0.f); x01 = fmaxf(x01, 0.f);
            x10 = fmaxf(x10, 0.f); x11 = fmaxf(x11, 0.f);
        }
        if (Cf) { Cf[o00] = x00; Cf[o01] = x01; Cf[o10] = x10; Cf[o11] = x11; }
        if (Cb) { Cb[o00] = f2bf(x00); Cb[o01] = f2bf(x01);
                  Cb[o10] = f2bf(x10); Cb[o11] = f2bf(x11); }
    }
}

// ---------------------------------------------------------------------------
// LayerNorm over a row of width Wd (256 or 512), optional residual add,
// writes f32 and/or bf16 outputs. One row per 128-thread block.
// ---------------------------------------------------------------------------
__global__ __launch_bounds__(128) void k_ln(
    const float* __restrict__ X, const float* __restrict__ R,
    const float* __restrict__ g, const float* __restrict__ bb,
    float* __restrict__ Of, unsigned short* __restrict__ Ob, int Wd)
{
    __shared__ float red[128];
    const int row = blockIdx.x, t = threadIdx.x;
    const size_t base = (size_t)row * Wd;
    const int nper = Wd >> 7;
    float v[4];
    float s = 0.f;
    for (int i = 0; i < nper; ++i) {
        int c = t + (i << 7);
        float x = X[base + c];
        if (R) x += R[base + c];
        v[i] = x; s += x;
    }
    red[t] = s; __syncthreads();
    for (int off = 64; off; off >>= 1) { if (t < off) red[t] += red[t + off]; __syncthreads(); }
    const float mean = red[0] / (float)Wd;
    __syncthreads();
    float s2 = 0.f;
    for (int i = 0; i < nper; ++i) { float d = v[i] - mean; s2 += d * d; }
    red[t] = s2; __syncthreads();
    for (int off = 64; off; off >>= 1) { if (t < off) red[t] += red[t + off]; __syncthreads(); }
    const float rstd = rsqrtf(red[0] / (float)Wd + 1e-5f);
    for (int i = 0; i < nper; ++i) {
        int c = t + (i << 7);
        float o = (v[i] - mean) * rstd * g[c] + bb[c];
        if (Of) Of[base + c] = o;
        if (Ob) Ob[base + c] = f2bf(o);
    }
}

// ---------------------------------------------------------------------------
// Build src = cat(scatter(last_hid, mcur), cos(dis*w+b)); write f32 + bf16
// ---------------------------------------------------------------------------
__global__ void k_src(
    const float* __restrict__ last_hid, const float* __restrict__ dis,
    const float* __restrict__ dw, const float* __restrict__ db,
    const float* __restrict__ mcur, const int* __restrict__ idxb,
    float* __restrict__ X, unsigned short* __restrict__ Sb, long long total)
{
    long long t = (long long)blockIdx.x * blockDim.x + threadIdx.x;
    if (t >= total) return;
    const int d = (int)(t % 512);
    const long long bn = t / 512;
    const int n = (int)(bn % 5);
    const long long b = bn / 5;
    float val;
    if (d < 256) {
        val = (n == idxb[b]) ? mcur[b * 256 + d] : last_hid[(b * 5 + n) * 256 + d];
    } else {
        const int dd = d - 256;
        val = __cosf(dis[b * 5 + n] * dw[dd] + db[dd]);
    }
    X[t] = val;
    Sb[t] = f2bf(val);
}

// ---------------------------------------------------------------------------
// MHA core: wave per (batch, head); 5x5 scores, softmax, out. hd=128, bf16 io.
// ---------------------------------------------------------------------------
__global__ __launch_bounds__(256) void k_attn(
    const unsigned short* __restrict__ QKV, unsigned short* __restrict__ O)
{
    const int gw = (blockIdx.x * 256 + threadIdx.x) >> 5;  // wave id = b*4 + h
    const int b = gw >> 2, h = gw & 3;
    const int l = threadIdx.x & 31;
    const size_t rb = (size_t)b * 5 * 1536 + h * 128 + l * 4;

    float kf[5][4], vf[5][4];
    #pragma unroll
    for (int i = 0; i < 5; ++i) {
        uint2 kk = *(const uint2*)(QKV + rb + (size_t)i * 1536 + 512);
        uint2 vv = *(const uint2*)(QKV + rb + (size_t)i * 1536 + 1024);
        kf[i][0] = bf2f((unsigned short)(kk.x & 0xffff)); kf[i][1] = bf2f((unsigned short)(kk.x >> 16));
        kf[i][2] = bf2f((unsigned short)(kk.y & 0xffff)); kf[i][3] = bf2f((unsigned short)(kk.y >> 16));
        vf[i][0] = bf2f((unsigned short)(vv.x & 0xffff)); vf[i][1] = bf2f((unsigned short)(vv.x >> 16));
        vf[i][2] = bf2f((unsigned short)(vv.y & 0xffff)); vf[i][3] = bf2f((unsigned short)(vv.y >> 16));
    }
    const float scale = 0.08838834764831845f;  // 1/sqrt(128)
    #pragma unroll
    for (int i = 0; i < 5; ++i) {
        uint2 qq = *(const uint2*)(QKV + rb + (size_t)i * 1536);
        float q0 = bf2f((unsigned short)(qq.x & 0xffff)), q1 = bf2f((unsigned short)(qq.x >> 16));
        float q2 = bf2f((unsigned short)(qq.y & 0xffff)), q3 = bf2f((unsigned short)(qq.y >> 16));
        float s[5];
        #pragma unroll
        for (int j = 0; j < 5; ++j) {
            float p = q0 * kf[j][0] + q1 * kf[j][1] + q2 * kf[j][2] + q3 * kf[j][3];
            for (int off = 16; off; off >>= 1) p += __shfl_xor(p, off);
            s[j] = p * scale;
        }
        float m = s[0];
        #pragma unroll
        for (int j = 1; j < 5; ++j) m = fmaxf(m, s[j]);
        float e[5], se = 0.f;
        #pragma unroll
        for (int j = 0; j < 5; ++j) { e[j] = __expf(s[j] - m); se += e[j]; }
        const float inv = 1.f / se;
        float o0 = 0.f, o1 = 0.f, o2 = 0.f, o3 = 0.f;
        #pragma unroll
        for (int j = 0; j < 5; ++j) {
            float a = e[j] * inv;
            o0 += a * vf[j][0]; o1 += a * vf[j][1]; o2 += a * vf[j][2]; o3 += a * vf[j][3];
        }
        uint2 po;
        po.x = (unsigned)f2bf(o0) | ((unsigned)f2bf(o1) << 16);
        po.y = (unsigned)f2bf(o2) | ((unsigned)f2bf(o3) << 16);
        *(uint2*)(O + ((size_t)b * 5 + i) * 512 + h * 128 + l * 4) = po;
    }
}

// ---------------------------------------------------------------------------
// cat(src, enc) bf16 -> [B, 5120]
// ---------------------------------------------------------------------------
__global__ void k_cat(const unsigned short* __restrict__ a,
                      const unsigned short* __restrict__ b,
                      unsigned short* __restrict__ o, long long n)
{
    long long t = (long long)blockIdx.x * blockDim.x + threadIdx.x;
    if (t >= n) return;
    const long long row = t / 2560, c = t % 2560;
    o[row * 5120 + c] = a[t];
    o[row * 5120 + 2560 + c] = b[t];
}

// ---------------------------------------------------------------------------
extern "C" void kernel_launch(void* const* d_in, const int* in_sizes, int n_in,
                              void* d_out, int out_size, void* d_ws, size_t ws_size,
                              hipStream_t stream)
{
    (void)n_in; (void)out_size; (void)ws_size;
    const int B = in_sizes[0] / 128;            // 16384
    const int M5 = B * 5;

    const float* obs      = (const float*)d_in[0];
    const float* last_hid = (const float*)d_in[1];
    const float* dis      = (const float*)d_in[2];
    const float* m1fc1w = (const float*)d_in[3];  const float* m1fc1b = (const float*)d_in[4];
    const float* m1ln1g = (const float*)d_in[5];  const float* m1ln1b = (const float*)d_in[6];
    const float* m1fc2w = (const float*)d_in[7];  const float* m1fc2b = (const float*)d_in[8];
    const float* m1ln2g = (const float*)d_in[9];  const float* m1ln2b = (const float*)d_in[10];
    const float* disw   = (const float*)d_in[11]; const float* disb   = (const float*)d_in[12];
    const float* m2fc1w = (const float*)d_in[37]; const float* m2fc1b = (const float*)d_in[38];
    const float* m2ln1g = (const float*)d_in[39]; const float* m2ln1b = (const float*)d_in[40];
    const float* m2fc2w = (const float*)d_in[41]; const float* m2fc2b = (const float*)d_in[42];
    const float* m2ln2g = (const float*)d_in[43]; const float* m2ln2b = (const float*)d_in[44];

    // workspace layout (~850 MB)
    char* ws = (char*)d_ws;
    size_t off = 0;
    auto alloc = [&](size_t bytes) -> char* {
        char* p = ws + off; off = (off + bytes + 255) & ~(size_t)255; return p;
    };
    unsigned short* Wm1fc1 = (unsigned short*)alloc((size_t)98304 * 2);
    unsigned short* Wm1fc2 = (unsigned short*)alloc((size_t)65536 * 2);
    unsigned short *Wqkv[2], *Wout[2], *Wff1[2], *Wff2[2];
    for (int L = 0; L < 2; ++L) {
        Wqkv[L] = (unsigned short*)alloc((size_t)786432 * 2);
        Wout[L] = (unsigned short*)alloc((size_t)262144 * 2);
        Wff1[L] = (unsigned short*)alloc((size_t)524288 * 2);
        Wff2[L] = (unsigned short*)alloc((size_t)524288 * 2);
    }
    unsigned short* Wm2fc1 = (unsigned short*)alloc((size_t)2621440 * 2);
    unsigned short* Wm2fc2 = (unsigned short*)alloc((size_t)131072 * 2);
    int* idxb = (int*)alloc((size_t)B * 4);
    float* X  = (float*)alloc((size_t)M5 * 512 * 4);      // encoder activation f32
    float* Y  = (float*)alloc((size_t)M5 * 512 * 4);      // GEMM f32 out / residual branch
    unsigned short* SRCB = (unsigned short*)alloc((size_t)M5 * 512 * 2);
    unsigned short* XB   = (unsigned short*)alloc((size_t)M5 * 512 * 2);
    unsigned short* S0   = (unsigned short*)alloc((size_t)M5 * 1536 * 2); // qkv/ff1/cat/merge1 tmp
    unsigned short* S1   = (unsigned short*)alloc((size_t)M5 * 512 * 2);  // attn out / merge2 tmp
    // merge1 temporaries aliased into S0 (dead before encoder starts)
    unsigned short* A1 = S0;                                        // [B,384] bf16
    float* T  = (float*)((char*)S0 + (size_t)B * 384 * 2);          // [B,256] f32
    unsigned short* A2 = (unsigned short*)((char*)T + (size_t)B * 256 * 4);
    float* MC = (float*)((char*)A2 + (size_t)B * 256 * 2);          // [B,256] f32
    // merge2 temporaries aliased into S1 (attn dead by then)
    float* H1 = (float*)S1;                                         // [B,512] f32
    unsigned short* H1B = (unsigned short*)((char*)H1 + (size_t)B * 512 * 4);
    float* H2 = Y;                                                  // [B,256] f32

    auto cvt = [&](const float* s, unsigned short* d, int n) {
        k_cvt<<<(n + 255) / 256, 256, 0, stream>>>(s, d, n);
    };
    auto gemm = [&](const unsigned short* Ap, const unsigned short* Wp, const float* bp,
                    float* Cf, unsigned short* Cb, int M, int N, int K, int relu) {
        k_gemm<<<dim3((M / 32) * (N / 256)), 256, 0, stream>>>(Ap, Wp, bp, Cf, Cb, M, N, K, relu);
    };

    // 0) weights -> bf16
    cvt(m1fc1w, Wm1fc1, 98304);
    cvt(m1fc2w, Wm1fc2, 65536);
    for (int L = 0; L < 2; ++L) {
        const int pb = 13 + L * 12;
        cvt((const float*)d_in[pb + 0], Wqkv[L], 786432);
        cvt((const float*)d_in[pb + 2], Wout[L], 262144);
        cvt((const float*)d_in[pb + 6], Wff1[L], 524288);
        cvt((const float*)d_in[pb + 8], Wff2[L], 524288);
    }
    cvt(m2fc1w, Wm2fc1, 2621440);
    cvt(m2fc2w, Wm2fc2, 131072);

    // 1) merge1
    k_prep<<<B, 128, 0, stream>>>(obs, last_hid, dis, idxb, A1);
    gemm(A1, Wm1fc1, m1fc1b, T, nullptr, B, 256, 384, 1);
    k_ln<<<B, 128, 0, stream>>>(T, nullptr, m1ln1g, m1ln1b, nullptr, A2, 256);
    gemm(A2, Wm1fc2, m1fc2b, T, nullptr, B, 256, 256, 1);
    k_ln<<<B, 128, 0, stream>>>(T, nullptr, m1ln2g, m1ln2b, MC, nullptr, 256);

    // 2) scatter + distance embedding -> src (f32 + bf16)
    const long long tot = (long long)M5 * 512;
    k_src<<<(unsigned)((tot + 255) / 256), 256, 0, stream>>>(
        last_hid, dis, disw, disb, MC, idxb, X, SRCB, tot);

    // 3) encoder layers
    const unsigned short* xin = SRCB;
    for (int L = 0; L < 2; ++L) {
        const int pb = 13 + L * 12;
        const float* qkv_b = (const float*)d_in[pb + 1];
        const float* out_b = (const float*)d_in[pb + 3];
        const float* ln1g  = (const float*)d_in[pb + 4];
        const float* ln1b  = (const float*)d_in[pb + 5];
        const float* ff1b  = (const float*)d_in[pb + 7];
        const float* ff2b  = (const float*)d_in[pb + 9];
        const float* ln2g  = (const float*)d_in[pb + 10];
        const float* ln2b  = (const float*)d_in[pb + 11];

        gemm(xin, Wqkv[L], qkv_b, nullptr, S0, M5, 1536, 512, 0);       // QKV bf16
        k_attn<<<(B * 4) / 8, 256, 0, stream>>>(S0, S1);                // attn out bf16
        gemm(S1, Wout[L], out_b, Y, nullptr, M5, 512, 512, 0);          // proj f32
        k_ln<<<M5, 128, 0, stream>>>(Y, X, ln1g, ln1b, X, XB, 512);     // x=LN(x+proj)
        gemm(XB, Wff1[L], ff1b, nullptr, S0, M5, 1024, 512, 1);         // ff1 relu bf16
        gemm(S0, Wff2[L], ff2b, Y, nullptr, M5, 512, 1024, 0);          // ff2 f32
        k_ln<<<M5, 128, 0, stream>>>(Y, X, ln2g, ln2b, X, XB, 512);     // x=LN(x+ff)
        xin = XB;
    }

    // 4) merge2
    const long long ncat = (long long)B * 2560;
    k_cat<<<(unsigned)((ncat + 255) / 256), 256, 0, stream>>>(SRCB, XB, S0, ncat);
    gemm(S0, Wm2fc1, m2fc1b, H1, nullptr, B, 512, 5120, 1);
    k_ln<<<B, 128, 0, stream>>>(H1, nullptr, m2ln1g, m2ln1b, nullptr, H1B, 512);
    gemm(H1B, Wm2fc2, m2fc2b, H2, nullptr, B, 256, 512, 1);
    k_ln<<<B, 128, 0, stream>>>(H2, nullptr, m2ln2g, m2ln2b, (float*)d_out, nullptr, 256);
}
